// RelPosGraphAttention_2997887172854
// MI455X (gfx1250) — matrix operand order repaired and checked by reading, compiled, not run on hardware
//
#include <hip/hip_runtime.h>
#include <math.h>

// Transformer-XL relative attention, fused flash-style, bf16 WMMA on gfx1250.
#define NB  4
#define SQ  1024
#define HD  768
#define NHD 12
#define DHD 64
#define PS  2048

typedef __attribute__((ext_vector_type(8)))  float          fx8;
typedef __attribute__((ext_vector_type(16))) __bf16         bfx16;
typedef __attribute__((ext_vector_type(16))) unsigned short usx16;
typedef __attribute__((ext_vector_type(8)))  unsigned short usx8;

__device__ __forceinline__ unsigned short f2bfu(float f) {
  unsigned int u = __builtin_bit_cast(unsigned int, f);
  unsigned int r = u + 0x7fffu + ((u >> 16) & 1u);   // round-to-nearest-even
  return (unsigned short)(r >> 16);
}

__device__ __forceinline__ bfx16 pack16(usx8 lo, usx8 hi) {
  usx16 u;
#pragma unroll
  for (int i = 0; i < 8; ++i) { u[i] = lo[i]; u[8 + i] = hi[i]; }
  return __builtin_bit_cast(bfx16, u);
}

// A fragment: 16x32 bf16, source row-major [rows, ld].
// lane l: M = l&15 ; halves 0-7 -> K = kbase + k1 + 0..7, halves 8-15 -> K = kbase + 16 + k1 + 0..7
// with k1 = (l<16 ? 0 : 8)   (per CDNA5 ISA 7.12.2)
__device__ __forceinline__ bfx16 load_a_bf(const unsigned short* base, int ld, int kbase, int lane) {
  int row = lane & 15;
  int k1  = (lane & 16) ? 8 : 0;
  const unsigned short* p = base + (size_t)row * ld + kbase + k1;
  usx8 lo = *(const usx8*)(p);
  usx8 hi = *(const usx8*)(p + 16);
  return pack16(lo, hi);
}

// B fragment: 32x16 bf16, source row-major [Krows, ld], N contiguous.
// lane l: N = l&15 ; halves 0-7 -> K = kbase + k1 + 0..7, halves 8-15 -> K = kbase + 16 + k1 + 0..7
__device__ __forceinline__ bfx16 load_b_bf(const unsigned short* base, int ld, int kbase, int nbase, int lane) {
  int k1 = (lane & 16) ? 8 : 0;
  const unsigned short* p = base + (size_t)(kbase + k1) * ld + nbase + (lane & 15);
  usx16 u;
#pragma unroll
  for (int i = 0; i < 8; ++i) { u[i] = p[(size_t)i * ld]; u[8 + i] = p[(size_t)(16 + i) * ld]; }
  return __builtin_bit_cast(bfx16, u);
}

#define WMMA_BF16(A, B, C) \
  __builtin_amdgcn_wmma_f32_16x16x32_bf16(false, (A), false, (B), (short)0, (C), false, false)

// ---------------------------------------------------------------- convert
__global__ void relpos_cvt_bf16(const float* __restrict__ s, unsigned short* __restrict__ d, int n) {
  int i  = blockIdx.x * blockDim.x + threadIdx.x;
  int st = gridDim.x * blockDim.x;
  for (; i < n; i += st) d[i] = f2bfu(s[i]);
}

// ---------------------------------------------------------------- projections
// 32-row x 64-col tile per wave: each B (weight) fragment feeds 2 WMMAs.
// mode 0: out0 = X@W + r_w_bias, out1 = X@W + r_r_bias   (layout [B,NH,seq,64])
// mode 1: out0 = X@W [B,NH,seq,64],  out1 = transpose -> [B,NH,64,seq]
// mode 2: out1 = transpose only -> [B,NH,64,seq]
__global__ void __launch_bounds__(32)
relpos_proj(const unsigned short* __restrict__ X, const unsigned short* __restrict__ W,
            const float* __restrict__ biasw, const float* __restrict__ biasr,
            unsigned short* __restrict__ out0, unsigned short* __restrict__ out1,
            int seqLen, int mode) {
  const int lane = threadIdx.x;
  const int m0   = blockIdx.x * 32;
  const int h    = blockIdx.y;
  const unsigned short* Ab = X + (size_t)m0 * HD;

  fx8 C[2][4];
#pragma unroll
  for (int r = 0; r < 2; ++r)
#pragma unroll
    for (int dt = 0; dt < 4; ++dt) C[r][dt] = (fx8){};

  for (int kk = 0; kk < HD / 32; ++kk) {
    bfx16 a0 = load_a_bf(Ab, HD, kk * 32, lane);
    bfx16 a1 = load_a_bf(Ab + (size_t)16 * HD, HD, kk * 32, lane);
#pragma unroll
    for (int dt = 0; dt < 4; ++dt) {
      bfx16 bw = load_b_bf(W, HD, kk * 32, h * DHD + dt * 16, lane);
      C[0][dt] = WMMA_BF16(a0, bw, C[0][dt]);
      C[1][dt] = WMMA_BF16(a1, bw, C[1][dt]);
    }
  }

  const int rowOff = (lane & 16) ? 8 : 0;
  const int jj     = lane & 15;
  const int bidx   = m0 / seqLen;
  const int ib     = m0 % seqLen;

#pragma unroll
  for (int r = 0; r < 2; ++r) {
#pragma unroll
    for (int dt = 0; dt < 4; ++dt) {
#pragma unroll
      for (int v = 0; v < 8; ++v) {
        int   d   = dt * 16 + jj;
        int   row = ib + r * 16 + v + rowOff;
        float c   = C[r][dt][v];
        if (mode == 0) {
          size_t o = ((size_t)(bidx * NHD + h) * SQ + row) * DHD + d;
          out0[o] = f2bfu(c + biasw[h * DHD + d]);
          out1[o] = f2bfu(c + biasr[h * DHD + d]);
        } else if (mode == 1) {
          size_t o  = ((size_t)(bidx * NHD + h) * SQ + row) * DHD + d;
          size_t ot = ((size_t)((bidx * NHD + h) * DHD + d)) * SQ + row;
          out0[o]  = f2bfu(c);
          out1[ot] = f2bfu(c);
        } else {
          size_t ot = ((size_t)((bidx * NHD + h) * DHD + d)) * PS + row;
          out1[ot] = f2bfu(c);
        }
      }
    }
  }
}

// ---------------------------------------------------------------- fused flash attention
__global__ void __launch_bounds__(32)
relpos_attn(const unsigned short* __restrict__ qw, const unsigned short* __restrict__ qr,
            const unsigned short* __restrict__ kv_n, const unsigned short* __restrict__ kv_t,
            const unsigned short* __restrict__ r_t, const float* __restrict__ mask,
            unsigned short* __restrict__ res_bf) {
  __shared__ float lds[16 * 32];

  const int    lane = threadIdx.x;
  const int    i0   = blockIdx.x * 16;
  const int    h    = blockIdx.y;
  const int    b    = blockIdx.z;
  const size_t bh   = (size_t)b * NHD + h;

  const unsigned short* qwb  = qw + (bh * SQ + i0) * DHD;
  const unsigned short* qrb  = qr + (bh * SQ + i0) * DHD;
  const unsigned short* kvtb = kv_t + bh * (size_t)DHD * SQ;
  const unsigned short* rtb  = r_t + bh * (size_t)DHD * PS;
  const unsigned short* kvnb = kv_n + bh * (size_t)SQ * DHD;
  const float*          mkb  = mask + (size_t)b * SQ * SQ;

  const bfx16 qwA0 = load_a_bf(qwb, DHD, 0, lane);
  const bfx16 qwA1 = load_a_bf(qwb, DHD, 32, lane);
  const bfx16 qrA0 = load_a_bf(qrb, DHD, 0, lane);
  const bfx16 qrA1 = load_a_bf(qrb, DHD, 32, lane);

  fx8 O[4];
#pragma unroll
  for (int dt = 0; dt < 4; ++dt) O[dt] = (fx8){};
  float m_[8], l_[8];
#pragma unroll
  for (int v = 0; v < 8; ++v) { m_[v] = -1.0e30f; l_[v] = 0.0f; }

  const int rowOff = (lane & 16) ? 8 : 0;
  const int jj     = lane & 15;
  const int k1     = (lane & 16) ? 8 : 0;

  // bd boundary-tile carry: pPrev always holds qr . r_head over columns [c0-16, c0)
  // of the *next* tile to be processed. Init for j0=0,t=0: columns [S-i0-16, S-i0).
  fx8 pPrev = (fx8){};
  {
    const int ci = SQ - i0 - 16;  // >= 0 for all i0
    pPrev = WMMA_BF16(qrA0, load_b_bf(rtb, PS, 0, ci, lane), pPrev);
    pPrev = WMMA_BF16(qrA1, load_b_bf(rtb, PS, 32, ci, lane), pPrev);
  }

  for (int j0 = 0; j0 < SQ; j0 += 32) {
    if (j0 + 32 < SQ) __builtin_prefetch(kvnb + (size_t)(j0 + 32 + lane) * DHD, 0, 1);

    fx8 sc[2];
#pragma unroll
    for (int t = 0; t < 2; ++t) {
      const int jb = j0 + 16 * t;
      // ac = (q+rw_bias) . kv^T
      fx8 acc = (fx8){};
      acc = WMMA_BF16(qwA0, load_b_bf(kvtb, SQ, 0, jb, lane), acc);
      acc = WMMA_BF16(qwA1, load_b_bf(kvtb, SQ, 32, jb, lane), acc);
      // bd band: PL = carried pPrev (cols [c0-16,c0)), PH = new tile (cols [c0,c0+16))
      const int c0 = SQ + jb - i0;  // 16-aligned
      fx8 pCur = (fx8){};
      pCur = WMMA_BF16(qrA0, load_b_bf(rtb, PS, 0, c0, lane), pCur);
      pCur = WMMA_BF16(qrA1, load_b_bf(rtb, PS, 32, c0, lane), pCur);
      // rel_shift via LDS skew read: bd(ii,jj) = band(ii, 16 + jj - ii)
      __syncthreads();
#pragma unroll
      for (int v = 0; v < 8; ++v) {
        int Mr = v + rowOff;
        lds[Mr * 32 + jj]      = pPrev[v];
        lds[Mr * 32 + 16 + jj] = pCur[v];
      }
      __syncthreads();
#pragma unroll
      for (int v = 0; v < 8; ++v) {
        int   Mr = v + rowOff;
        float bd = lds[Mr * 32 + 16 + jj - Mr];
        float mk = mkb[(size_t)(i0 + Mr) * SQ + jb + jj];
        sc[t][v] = (acc[v] + bd) * 0.125f + mk * (-65500.0f);
      }
      pPrev = pCur;
    }

    // online softmax over this 32-column chunk
    fx8 pt[2];
#pragma unroll
    for (int v = 0; v < 8; ++v) {
      float rm = fmaxf(sc[0][v], sc[1][v]);
#pragma unroll
      for (int off = 8; off >= 1; off >>= 1) rm = fmaxf(rm, __shfl_xor(rm, off, 32));
      float mnew  = fmaxf(m_[v], rm);
      float scale = __expf(m_[v] - mnew);
      m_[v]       = mnew;
      float p0 = __expf(sc[0][v] - mnew);
      float p1 = __expf(sc[1][v] - mnew);
      float rs = p0 + p1;
#pragma unroll
      for (int off = 8; off >= 1; off >>= 1) rs += __shfl_xor(rs, off, 32);
      l_[v] = l_[v] * scale + rs;
      O[0][v] *= scale; O[1][v] *= scale; O[2][v] *= scale; O[3][v] *= scale;
      pt[0][v] = p0; pt[1][v] = p1;
    }

    // route P (C layout) through LDS into A-fragment layout
    __syncthreads();
#pragma unroll
    for (int v = 0; v < 8; ++v) {
      int Mr = v + rowOff;
      lds[Mr * 32 + jj]      = pt[0][v];
      lds[Mr * 32 + 16 + jj] = pt[1][v];
    }
    __syncthreads();
    const float* pr = lds + (size_t)(lane & 15) * 32;
    usx16 u;
#pragma unroll
    for (int i = 0; i < 8; ++i) {
      u[i]     = f2bfu(pr[k1 + i]);
      u[8 + i] = f2bfu(pr[16 + k1 + i]);
    }
    bfx16 pA = __builtin_bit_cast(bfx16, u);

    // O += P @ V  (K = 32 columns of this chunk)
#pragma unroll
    for (int dt = 0; dt < 4; ++dt) {
      bfx16 vb = load_b_bf(kvnb, DHD, j0, dt * 16, lane);
      O[dt] = WMMA_BF16(pA, vb, O[dt]);
    }
  }

  // normalize + store per-head slice of res (bf16, [B*S, 768])
#pragma unroll
  for (int v = 0; v < 8; ++v) {
    float inv = 1.0f / l_[v];
#pragma unroll
    for (int dt = 0; dt < 4; ++dt) {
      float  o   = O[dt][v] * inv;
      size_t idx = (size_t)(b * SQ + i0 + v + rowOff) * HD + h * DHD + dt * 16 + jj;
      res_bf[idx] = f2bfu(o);
    }
  }
}

// ---------------------------------------------------------------- out-proj + residual + GELU
// 32-row x 64-col tile per wave (B fragments reused by 2 WMMAs)
__global__ void __launch_bounds__(32)
relpos_outproj(const unsigned short* __restrict__ resb, const unsigned short* __restrict__ wob,
               const float* __restrict__ b_out, const float* __restrict__ hidden,
               float* __restrict__ out) {
  const int lane = threadIdx.x;
  const int m0   = blockIdx.x * 32;
  const int nb   = blockIdx.y * 64;
  const unsigned short* Ab = resb + (size_t)m0 * HD;

  fx8 C[2][4];
#pragma unroll
  for (int r = 0; r < 2; ++r)
#pragma unroll
    for (int dt = 0; dt < 4; ++dt) C[r][dt] = (fx8){};

  for (int kk = 0; kk < HD / 32; ++kk) {
    bfx16 a0 = load_a_bf(Ab, HD, kk * 32, lane);
    bfx16 a1 = load_a_bf(Ab + (size_t)16 * HD, HD, kk * 32, lane);
#pragma unroll
    for (int dt = 0; dt < 4; ++dt) {
      bfx16 bw = load_b_bf(wob, HD, kk * 32, nb + dt * 16, lane);
      C[0][dt] = WMMA_BF16(a0, bw, C[0][dt]);
      C[1][dt] = WMMA_BF16(a1, bw, C[1][dt]);
    }
  }

  const int rowOff = (lane & 16) ? 8 : 0;
  const int jj     = lane & 15;
#pragma unroll
  for (int r = 0; r < 2; ++r) {
#pragma unroll
    for (int dt = 0; dt < 4; ++dt) {
#pragma unroll
      for (int v = 0; v < 8; ++v) {
        int    n   = nb + dt * 16 + jj;
        size_t idx = (size_t)(m0 + r * 16 + v + rowOff) * HD + n;
        float  z   = C[r][dt][v] + b_out[n] + hidden[idx];
        out[idx]   = 0.5f * z * (1.0f + erff(z * 0.70710678118654752f));  // exact GELU
      }
    }
  }
}

// ---------------------------------------------------------------- host
extern "C" void kernel_launch(void* const* d_in, const int* in_sizes, int n_in,
                              void* d_out, int out_size, void* d_ws, size_t ws_size,
                              hipStream_t stream) {
  const float* hidden  = (const float*)d_in[0];  // (B,S,H)
  const float* pos_emb = (const float*)d_in[1];  // (B,POS,H)
  const float* mask    = (const float*)d_in[2];  // (B,S,S)
  const float* qW      = (const float*)d_in[3];  // (H,NH,DH)
  const float* kvW     = (const float*)d_in[4];
  const float* rW      = (const float*)d_in[5];
  const float* r_r_b   = (const float*)d_in[6];  // (NH,DH)
  const float* r_w_b   = (const float*)d_in[7];
  const float* w_out   = (const float*)d_in[8];  // (H,H)
  const float* b_out   = (const float*)d_in[9];
  float*       out     = (float*)d_out;

  // workspace carve-up (bf16 stored as u16), all 256B aligned
  char*  ws  = (char*)d_ws;
  size_t off = 0;
  auto carve = [&](size_t nelem) {
    unsigned short* p = (unsigned short*)(ws + off);
    off += ((nelem * 2 + 255) & ~(size_t)255);
    return p;
  };
  unsigned short* hid_bf  = carve((size_t)NB * SQ * HD);
  unsigned short* pos_bf  = carve((size_t)NB * PS * HD);
  unsigned short* q_bf    = carve((size_t)HD * HD);
  unsigned short* kv_bf   = carve((size_t)HD * HD);
  unsigned short* r_bf    = carve((size_t)HD * HD);
  unsigned short* wo_bf   = carve((size_t)HD * HD);
  unsigned short* qw_h    = carve((size_t)NB * NHD * SQ * DHD);
  unsigned short* qr_h    = carve((size_t)NB * NHD * SQ * DHD);
  unsigned short* kv_nrm  = carve((size_t)NB * NHD * SQ * DHD);
  unsigned short* kv_trs  = carve((size_t)NB * NHD * DHD * SQ);
  unsigned short* r_trs   = carve((size_t)NB * NHD * DHD * PS);
  unsigned short* res_bf  = carve((size_t)NB * SQ * HD);
  (void)ws_size; (void)in_sizes; (void)n_in; (void)out_size;

  auto cvt = [&](const float* s, unsigned short* d, int n) {
    int blk = 256, grd = (n + blk * 8 - 1) / (blk * 8);
    relpos_cvt_bf16<<<grd, blk, 0, stream>>>(s, d, n);
  };
  cvt(hidden, hid_bf, NB * SQ * HD);
  cvt(pos_emb, pos_bf, NB * PS * HD);
  cvt(qW, q_bf, HD * HD);
  cvt(kvW, kv_bf, HD * HD);
  cvt(rW, r_bf, HD * HD);
  cvt(w_out, wo_bf, HD * HD);

  // projections (32-row tiles)
  relpos_proj<<<dim3(NB * SQ / 32, NHD), 32, 0, stream>>>(
      hid_bf, q_bf, r_w_b, r_r_b, qw_h, qr_h, SQ, 0);
  relpos_proj<<<dim3(NB * SQ / 32, NHD), 32, 0, stream>>>(
      hid_bf, kv_bf, nullptr, nullptr, kv_nrm, kv_trs, SQ, 1);
  relpos_proj<<<dim3(NB * PS / 32, NHD), 32, 0, stream>>>(
      pos_bf, r_bf, nullptr, nullptr, nullptr, r_trs, PS, 2);

  // fused flash attention with rel_shift (bd boundary-tile carry)
  relpos_attn<<<dim3(SQ / 16, NHD, NB), 32, 0, stream>>>(
      qw_h, qr_h, kv_nrm, kv_trs, r_trs, mask, res_bf);

  // output projection + residual + exact GELU (32-row tiles)
  relpos_outproj<<<dim3(NB * SQ / 32, HD / 64), 32, 0, stream>>>(
      res_bf, wo_bf, b_out, hidden, out);
}
